// LSTM_11811160064747
// MI455X (gfx1250) — matrix operand order repaired and checked by reading, not compiled, not run on hardware
//
#include <hip/hip_runtime.h>
#include <hip/hip_bf16.h>

typedef _Float16 h16;
typedef __attribute__((ext_vector_type(16))) _Float16 v16h;
typedef __attribute__((ext_vector_type(8)))  _Float16 v8h;
typedef __attribute__((ext_vector_type(8)))  float    v8f;

#define DEVINL __device__ inline __attribute__((always_inline))

// ---------------- LDS layout (bytes), all offsets 32B aligned ----------------
// Combined, K-padded f16 weights (B-matrix source, row = gate, padded K cols).
// Biases are folded into a spare K column against a constant-1.0 activation:
//   WC1: 256 x 96  (0..14 = Wih1, 15 = bih1+bhh1, 16..31 = 0, 32..95 = Whh1)
//   WC2: 128 x 96  (0..63 = Wih2, 64..95 = Whh2)            [bias via splat]
//   WC3:  64 x 64  (0..31 = Wih3, 32..47 = Whh3, 48 = bias, 49..63 = 0)
//   WF1:  80 x 96  (0..79 = W1, 80 = b1, 81..95 = 0)
#define OFF_WC1   0        // 49152 B
#define OFF_WC2   49152    // 24576 B
#define OFF_WC3   73728    //  8192 B
#define OFF_WF1   81920    // 15360 B
#define OFF_BG2   97280    // 128 f32 (bih2+bhh2)
#define OFF_W2    97792    //  80 f32
#define OFF_BFC2  98112    //   1 f32 (+pad)
#define OFF_WAVE  98144    // per-wave scratch
// per-wave scratch (halfs): staged[16x64]=1024, h1[16x64]=1024,
//                           h2[16x32]=512, h3[16x32 padded]=512, fcbuf[16x96]=1536
#define WAVE_STRIDE 9216
#define NWAVES 8
#define SMEM_BYTES (OFF_WAVE + NWAVES * WAVE_STRIDE)   // 171872 B < 320 KB/WGP

// ---------------- math helpers ----------------
DEVINL float sig_(float x)  { return 1.0f / (1.0f + __expf(-x)); }
DEVINL float tanh_(float x) {
  float a = fabsf(x);
  float e = __expf(-2.0f * a);
  float r = (1.0f - e) / (1.0f + e);
  return copysignf(r, x);
}

union AFrag { v16h v; v8h p[2]; h16 e[16]; };

// B fragment (32x16 f16, K x N): lane<16 holds K=0..7,16..23 col N=lane,
// lane>=16 holds K=8..15,24..31 col N=lane-16. With W stored row-major [gate][Kpad],
// B[k][n] = W[n][k], so each lane reads two runs of 8 halfs (16B aligned).
DEVINL v16h ldsB(const h16* w, int Kpad, int n, int c, int lane) {
  const h16* p = w + (n * 16 + (lane & 15)) * Kpad + c * 32 + ((lane >> 4) & 1) * 8;
  AFrag f;
  f.p[0] = *(const v8h*)p;
  f.p[1] = *(const v8h*)(p + 16);
  return f.v;
}

// A fragment (16x32 f16, M x K) from an activation buffer act[m][Kpad]:
// lane<16: M=lane, halfs 0..7 -> K=kb+0..7,  halfs 8..15 -> K=kb+16..23
// lane>=16: M=lane-16, halfs 0..7 -> K=kb+8..15, halfs 8..15 -> K=kb+24..31
DEVINL v16h ldsA(const h16* act, int Kpad, int kbase, int lane) {
  const h16* p = act + (lane & 15) * Kpad + kbase + ((lane >> 4) & 1) * 8;
  AFrag f;
  f.p[0] = *(const v8h*)p;
  f.p[1] = *(const v8h*)(p + 16);
  return f.v;
}

// A fragment for layer-1 x input straight from global.
// K=0..14 from x, K=15 = 1.0 (bias column), K>=16 = 0.
DEVINL v16h load_x_frag(const float* __restrict__ x, long rowBase, int t, int lane, int Bn) {
  AFrag f;
  long m = rowBase + (lane & 15);
  if (m > (long)Bn - 1) m = (long)Bn - 1;          // clamp (keeps EXEC uniform)
  const float* row = x + m * 75 + t * 15;          // [B,T=5,I=15]
  int kadd = ((lane >> 4) & 1) * 8;
#pragma unroll
  for (int h = 0; h < 16; ++h) {
    int k = (h < 8 ? h : h + 8) + kadd;
    f.e[h] = (k < 15) ? (h16)row[k] : ((k == 15) ? (h16)1.0f : (h16)0.0f);
  }
  return f.v;
}

DEVINL v8f wmma32(v16h a, v16h b, v8f c) {
  return __builtin_amdgcn_wmma_f32_16x16x32_f16(false, a, false, b, (short)0, c, false, false);
}

// One LSTM timestep for a 16-row batch tile. Gates laid out PyTorch-style
// [i;f;g;o], gate tile n = ty*(H/16) + j. Cell state lives in registers.
// HAS_BIAS=false: bias folded into weights, acc starts at inline-0 C.
template <int H, int NC, bool HAS_BIAS>
DEVINL void lstm_step(const h16* Wl, int KpadW, const float* bias,
                      const v16h* afr, v8f* cst,
                      h16* hout, int hKpad,
                      h16* yout, int yKpad, int ycol, bool do_tanh, int lane) {
  const int HT = H / 16;
  const int N  = lane & 15;
  const int mh = ((lane >> 4) & 1) * 8;
#pragma unroll
  for (int j = 0; j < HT; ++j) {
    v8f acc[4];
#pragma unroll
    for (int ty = 0; ty < 4; ++ty) {
      int n = ty * HT + j;
      v8f a = {0.f, 0.f, 0.f, 0.f, 0.f, 0.f, 0.f, 0.f};
      if (HAS_BIAS) {
        float bval = bias[n * 16 + N];
#pragma unroll
        for (int e = 0; e < 8; ++e) a[e] = bval;
      }
#pragma unroll
      for (int c = 0; c < NC; ++c)
        a = wmma32(afr[c], ldsB(Wl, KpadW, n, c, lane), a);
      acc[ty] = a;
    }
    v8f cs = cst[j];
#pragma unroll
    for (int e = 0; e < 8; ++e) {
      float iv = sig_(acc[0][e]);
      float fv = sig_(acc[1][e]);
      float gv = tanh_(acc[2][e]);
      float ov = sig_(acc[3][e]);
      float cn = fv * cs[e] + iv * gv;
      float hv = ov * tanh_(cn);
      cs[e] = cn;
      int m = e + mh;
      hout[m * hKpad + j * 16 + N] = (h16)hv;
      yout[m * yKpad + ycol + j * 16 + N] = (h16)(do_tanh ? tanh_(hv) : hv);
    }
    cst[j] = cs;
  }
}

__global__ __launch_bounds__(256, 1)
void lstm3_fc_fused(const float* __restrict__ x,
                    const float* __restrict__ Wih1, const float* __restrict__ Whh1,
                    const float* __restrict__ bih1, const float* __restrict__ bhh1,
                    const float* __restrict__ Wih2, const float* __restrict__ Whh2,
                    const float* __restrict__ bih2, const float* __restrict__ bhh2,
                    const float* __restrict__ Wih3, const float* __restrict__ Whh3,
                    const float* __restrict__ bih3, const float* __restrict__ bhh3,
                    const float* __restrict__ W1,   const float* __restrict__ b1,
                    const float* __restrict__ W2,   const float* __restrict__ b2,
                    float* __restrict__ out, int Bn) {
  extern __shared__ char smem[];
  h16* WC1 = (h16*)(smem + OFF_WC1);
  h16* WC2 = (h16*)(smem + OFF_WC2);
  h16* WC3 = (h16*)(smem + OFF_WC3);
  h16* WF1 = (h16*)(smem + OFF_WF1);
  float* bg2  = (float*)(smem + OFF_BG2);
  float* w2   = (float*)(smem + OFF_W2);
  float* bfc2 = (float*)(smem + OFF_BFC2);

  const int tid  = threadIdx.x;
  const int lane = tid & 31;
  const int wave = tid >> 5;

  // ---- cooperative weight staging: f32 global -> padded-row f16 LDS ----
  for (int i = tid; i < 256 * 96; i += 256) {
    int g = i / 96, k = i - g * 96;
    float v = 0.0f;
    if (k < 15)        v = Wih1[g * 15 + k];
    else if (k == 15)  v = bih1[g] + bhh1[g];          // bias column
    else if (k >= 32)  v = Whh1[g * 64 + (k - 32)];
    WC1[i] = (h16)v;
  }
  for (int i = tid; i < 128 * 96; i += 256) {
    int g = i / 96, k = i - g * 96;
    float v = (k < 64) ? Wih2[g * 64 + k] : Whh2[g * 32 + (k - 64)];
    WC2[i] = (h16)v;
  }
  for (int i = tid; i < 64 * 64; i += 256) {
    int g = i >> 6, k = i & 63;
    float v = 0.0f;
    if (k < 32)       v = Wih3[g * 32 + k];
    else if (k < 48)  v = Whh3[g * 16 + (k - 32)];
    else if (k == 48) v = bih3[g] + bhh3[g];           // bias column
    WC3[i] = (h16)v;
  }
  for (int i = tid; i < 80 * 96; i += 256) {
    int g = i / 96, k = i - g * 96;
    float v = 0.0f;
    if (k < 80)       v = W1[g * 80 + k];
    else if (k == 80) v = b1[g];                       // bias column
    WF1[i] = (h16)v;
  }
  if (tid < 128) bg2[tid] = bih2[tid] + bhh2[tid];
  if (tid < 80)  w2[tid] = W2[tid];
  if (tid == 0)  bfc2[0] = b2[0];

  // ---- per-wave activation scratch ----
  h16* staged = (h16*)(smem + OFF_WAVE + wave * WAVE_STRIDE); // 16x64
  h16* h1s = staged + 1024;   // 16x64
  h16* h2s = staged + 2048;   // 16x32
  h16* h3s = staged + 2560;   // 16x32 (col 16 = 1.0 bias, 17..31 = 0)
  h16* fcb = staged + 3072;   // 16x96 (col 80 = 1.0 bias, 81..95 = 0)
  for (int i = lane; i < 3584; i += 32) staged[1024 + i] = (h16)0.0f;
  if (lane < 16) {
    h3s[lane * 32 + 16] = (h16)1.0f;   // constant-1 column vs WC3 col 48
    fcb[lane * 96 + 80] = (h16)1.0f;   // constant-1 column vs WF1 col 80
  }

  __syncthreads();

  const long rb = ((long)blockIdx.x * NWAVES + wave) * 16;

  v8f c1[4], c2s[2], c3s[1];
#pragma unroll
  for (int j = 0; j < 4; ++j)
#pragma unroll
    for (int e = 0; e < 8; ++e) c1[j][e] = 0.0f;
#pragma unroll
  for (int j = 0; j < 2; ++j)
#pragma unroll
    for (int e = 0; e < 8; ++e) c2s[j][e] = 0.0f;
#pragma unroll
  for (int e = 0; e < 8; ++e) c3s[0][e] = 0.0f;

#pragma unroll 1
  for (int t = 0; t < 5; ++t) {
    // layer 1: K = [x(15)|1|pad | h1(64)], 16 gate tiles x 3 k-chunks
    v16h a1[3];
    a1[0] = load_x_frag(x, rb, t, lane, Bn);
    a1[1] = ldsA(h1s, 64, 0, lane);
    a1[2] = ldsA(h1s, 64, 32, lane);
    lstm_step<64, 3, false>(WC1, 96, nullptr, a1, c1, h1s, 64, staged, 64, 0, true, lane);

    // layer 2: K = [y1(64) | h2(32)], 8 gate tiles x 3 k-chunks (bias splat)
    v16h a2[3];
    a2[0] = ldsA(staged, 64, 0, lane);
    a2[1] = ldsA(staged, 64, 32, lane);
    a2[2] = ldsA(h2s, 32, 0, lane);
    lstm_step<32, 3, true>(WC2, 96, bg2, a2, c2s, h2s, 32, staged, 64, 0, true, lane);

    // layer 3: K = [y2(32) | h3(16)|1|pad], 4 gate tiles x 2 k-chunks
    v16h a3[2];
    a3[0] = ldsA(staged, 64, 0, lane);
    a3[1] = ldsA(h3s, 32, 0, lane);
    lstm_step<16, 2, false>(WC3, 64, nullptr, a3, c3s, h3s, 32, fcb, 96, t * 16, false, lane);
  }

  // ---- FC1: [16x80] = relu(fc_in[16x80|1] @ [W1|b1]^T), 5 N-tiles x 3 k-chunks
  h16* fcout = h1s;  // reuse dead h1+h2 region (1536 halfs >= 16*80)
  v16h af[3];
  af[0] = ldsA(fcb, 96, 0, lane);
  af[1] = ldsA(fcb, 96, 32, lane);
  af[2] = ldsA(fcb, 96, 64, lane);
  const int N  = lane & 15;
  const int mh = ((lane >> 4) & 1) * 8;
#pragma unroll
  for (int n = 0; n < 5; ++n) {
    v8f a = {0.f, 0.f, 0.f, 0.f, 0.f, 0.f, 0.f, 0.f};
#pragma unroll
    for (int c = 0; c < 3; ++c)
      a = wmma32(af[c], ldsB(WF1, 96, n, c, lane), a);
#pragma unroll
    for (int e = 0; e < 8; ++e)
      fcout[(e + mh) * 80 + n * 16 + N] = (h16)fmaxf(a[e], 0.0f);
  }

  // ---- FC2: 80 -> 1, one row per lane (lanes 0..15)
  if (lane < 16) {
    long m = rb + lane;
    if (m < (long)Bn) {
      float s = bfc2[0];
#pragma unroll 1
      for (int f = 0; f < 80; ++f)
        s += (float)fcout[lane * 80 + f] * w2[f];
      out[m] = s;
    }
  }
}

extern "C" void kernel_launch(void* const* d_in, const int* in_sizes, int n_in,
                              void* d_out, int out_size, void* d_ws, size_t ws_size,
                              hipStream_t stream) {
  (void)n_in; (void)out_size; (void)d_ws; (void)ws_size;
  const float* x    = (const float*)d_in[0];
  const float* Wih1 = (const float*)d_in[1];
  const float* Whh1 = (const float*)d_in[2];
  const float* bih1 = (const float*)d_in[3];
  const float* bhh1 = (const float*)d_in[4];
  const float* Wih2 = (const float*)d_in[5];
  const float* Whh2 = (const float*)d_in[6];
  const float* bih2 = (const float*)d_in[7];
  const float* bhh2 = (const float*)d_in[8];
  const float* Wih3 = (const float*)d_in[9];
  const float* Whh3 = (const float*)d_in[10];
  const float* bih3 = (const float*)d_in[11];
  const float* bhh3 = (const float*)d_in[12];
  const float* W1   = (const float*)d_in[13];
  const float* b1   = (const float*)d_in[14];
  const float* W2   = (const float*)d_in[15];
  const float* b2   = (const float*)d_in[16];

  int Bn = in_sizes[0] / 75;                 // B*T*I / (T*I)
  int grid = (Bn + 127) / 128;               // 8 waves x 16 rows per block

  (void)hipFuncSetAttribute((const void*)lstm3_fc_fused,
                            hipFuncAttributeMaxDynamicSharedMemorySize,
                            (int)SMEM_BYTES);

  lstm3_fc_fused<<<grid, 256, SMEM_BYTES, stream>>>(
      x, Wih1, Whh1, bih1, bhh1, Wih2, Whh2, bih2, bhh2,
      Wih3, Whh3, bih3, bhh3, W1, b1, W2, b2,
      (float*)d_out, Bn);
}
